// BlocoTransformer_2611340116269
// MI455X (gfx1250) — compile-verified, matrix operand order checked by reading, not run on hardware
//
#include <hip/hip_runtime.h>

// ---------------------------------------------------------------------------
// Types
// ---------------------------------------------------------------------------
typedef __bf16 bf16;
typedef __bf16 v16bf __attribute__((ext_vector_type(16)));
typedef float  v8f   __attribute__((ext_vector_type(8)));
typedef unsigned int v4u __attribute__((ext_vector_type(4)));
typedef int v4i __attribute__((ext_vector_type(4)));

#define WMMA_BF16(a, b, c) \
  __builtin_amdgcn_wmma_f32_16x16x32_bf16(false, (a), false, (b), (short)0, (c), false, false)

#ifndef __has_builtin
#define __has_builtin(x) 0
#endif

#if __has_builtin(__builtin_amdgcn_global_load_async_to_lds_b128)
#define HAVE_ASYNC_LDS 1
#else
#define HAVE_ASYNC_LDS 0
#endif

#if HAVE_ASYNC_LDS
typedef __attribute__((address_space(1))) v4i gv4i;   // global <4 x i32>
typedef __attribute__((address_space(3))) v4i lv4i;   // LDS    <4 x i32>
__device__ __forceinline__ void async_copy16(const bf16* g, bf16* l) {
  __builtin_amdgcn_global_load_async_to_lds_b128((gv4i*)g, (lv4i*)l, 0, 0);
}
__device__ __forceinline__ void wait_async() {
#if __has_builtin(__builtin_amdgcn_s_wait_asynccnt)
  __builtin_amdgcn_s_wait_asynccnt(0);
#else
  asm volatile("s_wait_asynccnt 0x0" ::: "memory");
#endif
}
#else
__device__ __forceinline__ void wait_async() {}
#endif

// Load a 16x32 bf16 A/B fragment for one lane: two 16-byte chunks
// (K = kh+0..7 and K = kh+16..23 per the CDNA5 16-bit operand layout).
__device__ __forceinline__ v16bf load_frag(const bf16* p0, const bf16* p1) {
  union { v4u q[2]; v16bf v; } u;
  u.q[0] = *(const v4u*)p0;
  u.q[1] = *(const v4u*)p1;
  return u.v;
}

// ---------------------------------------------------------------------------
// Weight preparation
// ---------------------------------------------------------------------------
__global__ void f32_to_bf16_kernel(const float* __restrict__ src,
                                   bf16* __restrict__ dst, int n) {
  int i = blockIdx.x * blockDim.x + threadIdx.x;
  if (i < n) dst[i] = (bf16)src[i];
}

// Pack Wq/Wk/Wv [H,C,DH] into one bf16 [C, 3*H*DH] matrix:
// out[c][s*H*DH + h*DH + d] = W_s[h][c][d]
__global__ void pack_qkv_kernel(const float* __restrict__ Wq,
                                const float* __restrict__ Wk,
                                const float* __restrict__ Wv,
                                bf16* __restrict__ out,
                                int H, int C, int DH) {
  int o = blockIdx.x * blockDim.x + threadIdx.x;
  int HD = H * DH;
  int n3 = 3 * HD;
  int total = C * n3;
  if (o >= total) return;
  int c = o / n3;
  int rem = o - c * n3;
  int s = rem / HD;
  int hd = rem - s * HD;
  int h = hd / DH;
  int d = hd - h * DH;
  const float* W = (s == 0) ? Wq : (s == 1) ? Wk : Wv;
  out[o] = (bf16)W[((size_t)h * C + c) * DH + d];
}

// ---------------------------------------------------------------------------
// LayerNorm -> bf16
// ---------------------------------------------------------------------------
__global__ void ln_bf16_kernel(const float* __restrict__ x,
                               const float* __restrict__ g,
                               const float* __restrict__ be,
                               bf16* __restrict__ out, int C) {
  int row = blockIdx.x;
  const float* xr = x + (size_t)row * C;
  float s = 0.f, ss = 0.f;
  for (int i = threadIdx.x; i < C; i += blockDim.x) {
    float v = xr[i];
    s += v; ss += v * v;
  }
#pragma unroll
  for (int o = 16; o > 0; o >>= 1) {
    s += __shfl_xor(s, o);
    ss += __shfl_xor(ss, o);
  }
  __shared__ float rs[8], rss[8];
  int wave = threadIdx.x >> 5, lane = threadIdx.x & 31;
  if (lane == 0) { rs[wave] = s; rss[wave] = ss; }
  __syncthreads();
  float S = 0.f, SS = 0.f;
#pragma unroll
  for (int i = 0; i < 8; ++i) { S += rs[i]; SS += rss[i]; }
  float mu = S / (float)C;
  float var = SS / (float)C - mu * mu;
  float inv = rsqrtf(var + 1e-5f);
  for (int i = threadIdx.x; i < C; i += blockDim.x)
    out[(size_t)row * C + i] = (bf16)((xr[i] - mu) * inv * g[i] + be[i]);
}

// ---------------------------------------------------------------------------
// bf16 WMMA GEMM: C[M,N] = A[M,K] * B[K,N] (+bias, relu, +residual)
// Block tile 128x128, BK=32, 256 threads = 8 waves. Double-buffered LDS,
// one barrier per K-step; next tile's global loads overlap the WMMAs.
// Wave w owns rows w*16..w*16+16 of the tile, all 128 columns (8 WMMAs/step).
// ---------------------------------------------------------------------------
__global__ __launch_bounds__(256)
void gemm_bf16_kernel(const bf16* __restrict__ A, const bf16* __restrict__ Bm,
                      const float* __restrict__ bias,
                      const float* __restrict__ resid,
                      float* __restrict__ outF, bf16* __restrict__ outB,
                      int M, int N, int K, int relu) {
  __shared__ __align__(16) bf16 As[2][128][40];  // padded rows keep 16B align
  __shared__ __align__(16) bf16 Bt[2][128][40];  // B tile transposed [n][k]

  const int tid = threadIdx.x;
  const int wave = tid >> 5, lane = tid & 31;
  const int ml = lane & 15;
  const int lh = (lane < 16) ? 0 : 8;
  const int rbase = (lane < 16) ? 0 : 8;
  const int m0 = blockIdx.y * 128;
  const int n0 = blockIdx.x * 128;

  v8f acc[8] = {};

  const int ar = tid >> 1, ac = (tid & 1) << 4;  // A: 2 threads/row, 16 elems
  const int bk = tid >> 3, bn = (tid & 7) << 4;  // B: 8 threads/row, 16 elems

  const int nk = K >> 5;

  // ---- prologue: stage tile 0 into buffer 0 ----
  {
    const bf16* ap = A + (size_t)(m0 + ar) * K + ac;
    const bf16* bp = Bm + (size_t)bk * N + n0 + bn;
#if HAVE_ASYNC_LDS
    async_copy16(ap, &As[0][ar][ac]);
    async_copy16(ap + 8, &As[0][ar][ac + 8]);
#else
    *(v4u*)&As[0][ar][ac]     = *(const v4u*)ap;
    *(v4u*)&As[0][ar][ac + 8] = *(const v4u*)(ap + 8);
#endif
    v4u b0 = *(const v4u*)bp;
    v4u b1 = *(const v4u*)(bp + 8);
    const bf16* e0 = (const bf16*)&b0;
    const bf16* e1 = (const bf16*)&b1;
#pragma unroll
    for (int j = 0; j < 8; ++j) {
      Bt[0][bn + j][bk]     = e0[j];
      Bt[0][bn + 8 + j][bk] = e1[j];
    }
    wait_async();
    __syncthreads();
  }

  for (int kt = 0; kt < nk; ++kt) {
    const int bi = kt & 1;
    const bool pre = (kt + 1 < nk);
    v4u nb0 = {}, nb1 = {};
#if !HAVE_ASYNC_LDS
    v4u na0 = {}, na1 = {};
#endif
    // issue next tile's global loads (latency hidden behind WMMAs below)
    if (pre) {
      const bf16* ap = A + (size_t)(m0 + ar) * K + ((kt + 1) << 5) + ac;
      const bf16* bp = Bm + (size_t)(((kt + 1) << 5) + bk) * N + n0 + bn;
#if HAVE_ASYNC_LDS
      async_copy16(ap, &As[bi ^ 1][ar][ac]);
      async_copy16(ap + 8, &As[bi ^ 1][ar][ac + 8]);
#else
      na0 = *(const v4u*)ap;
      na1 = *(const v4u*)(ap + 8);
#endif
      nb0 = *(const v4u*)bp;
      nb1 = *(const v4u*)(bp + 8);
    }

    // ---- compute: 1 A fragment, 8 B fragments, 8 WMMAs ----
    const int am = (wave << 4) + ml;
    v16bf af = load_frag(&As[bi][am][lh], &As[bi][am][lh + 16]);
    v16bf bfr[4];
#pragma unroll
    for (int g = 0; g < 2; ++g) {
#pragma unroll
      for (int j = 0; j < 4; ++j) {
        const int bc = ((g * 4 + j) << 4) + ml;
        bfr[j] = load_frag(&Bt[bi][bc][lh], &Bt[bi][bc][lh + 16]);
      }
#pragma unroll
      for (int j = 0; j < 4; ++j)
        acc[g * 4 + j] = WMMA_BF16(af, bfr[j], acc[g * 4 + j]);
    }

    // ---- scatter next tile into the other LDS buffer ----
    if (pre) {
#if !HAVE_ASYNC_LDS
      *(v4u*)&As[bi ^ 1][ar][ac]     = na0;
      *(v4u*)&As[bi ^ 1][ar][ac + 8] = na1;
#endif
      const bf16* e0 = (const bf16*)&nb0;
      const bf16* e1 = (const bf16*)&nb1;
#pragma unroll
      for (int j = 0; j < 8; ++j) {
        Bt[bi ^ 1][bn + j][bk]     = e0[j];
        Bt[bi ^ 1][bn + 8 + j][bk] = e1[j];
      }
    }
    wait_async();
    __syncthreads();
  }

  // epilogue: lane holds column n for 8 rows per accumulator
#pragma unroll
  for (int j = 0; j < 8; ++j) {
    const int n = n0 + (j << 4) + ml;
    const float bval = bias ? bias[n] : 0.f;
#pragma unroll
    for (int r = 0; r < 8; ++r) {
      const int m = m0 + (wave << 4) + rbase + r;
      float v = acc[j][r] + bval;
      if (relu) v = fmaxf(v, 0.f);
      if (resid) v += resid[(size_t)m * N + n];
      if (outF) outF[(size_t)m * N + n] = v;
      if (outB) outB[(size_t)m * N + n] = (bf16)v;
    }
  }
}

// ---------------------------------------------------------------------------
// Flash attention over packed QKV [B*T, 3072] bf16 (q|k|v, each h*64+d).
// Grid: (T/64, B*H), block 128 (4 waves). Wave w owns q rows qb+w*16..+16.
// ---------------------------------------------------------------------------
__global__ __launch_bounds__(128)
void attn_kernel(const bf16* __restrict__ qkv, bf16* __restrict__ out, int T) {
  const int S3 = 3072;
  __shared__ __align__(16) bf16 Vt[64][40];     // V tile transposed [d][s]
  __shared__ __align__(16) bf16 P[4][16][40];   // per-wave probabilities

  const int tid = threadIdx.x;
  const int wave = tid >> 5, lane = tid & 31;
  const int ml = lane & 15;
  const int lh = (lane < 16) ? 0 : 8;
  const int rbase = (lane < 16) ? 0 : 8;

  const int bh = blockIdx.y;
  const int b = bh >> 4, h = bh & 15;
  const int qb = blockIdx.x * 64;
  const int qw = qb + wave * 16;

  const bf16* base = qkv + (size_t)b * T * S3;

  // Q fragments: lane ml -> row qw+ml, two 32-wide d chunks
  const bf16* qp = base + (size_t)(qw + ml) * S3 + h * 64;
  v16bf qf[2];
#pragma unroll
  for (int c = 0; c < 2; ++c)
    qf[c] = load_frag(qp + c * 32 + lh, qp + c * 32 + lh + 16);

  v8f oacc[4] = {};
  float mrow[8], lrow[8];
#pragma unroll
  for (int r = 0; r < 8; ++r) { mrow[r] = -1e30f; lrow[r] = 0.f; }
  const float scale = 0.125f;  // DH^-0.5

  for (int sb = 0; sb < qb + 64; sb += 32) {
    // stage V tile transposed: [d][sj]
    {
      const int sj = tid >> 2, dq = (tid & 3) << 4;
      const bf16* vp = base + (size_t)(sb + sj) * S3 + 2048 + h * 64 + dq;
      v4u v0 = *(const v4u*)vp;
      v4u v1 = *(const v4u*)(vp + 8);
      const bf16* e0 = (const bf16*)&v0;
      const bf16* e1 = (const bf16*)&v1;
#pragma unroll
      for (int j = 0; j < 8; ++j) {
        Vt[dq + j][sj]     = e0[j];
        Vt[dq + 8 + j][sj] = e1[j];
      }
    }
    __syncthreads();

    // scores S[16 x 32] = Q (16x64) * K^T : load all 4 K frags, then 4 WMMAs
    v16bf kf[4];
#pragma unroll
    for (int j = 0; j < 2; ++j) {
      const bf16* kp = base + (size_t)(sb + j * 16 + ml) * S3 + 1024 + h * 64;
#pragma unroll
      for (int c = 0; c < 2; ++c)
        kf[j * 2 + c] = load_frag(kp + c * 32 + lh, kp + c * 32 + lh + 16);
    }
    v8f sacc[2] = {};
#pragma unroll
    for (int j = 0; j < 2; ++j)
#pragma unroll
      for (int c = 0; c < 2; ++c)
        sacc[j] = WMMA_BF16(qf[c], kf[j * 2 + c], sacc[j]);

    // causal mask + online softmax (row = across 16 lanes of the half-wave)
#pragma unroll
    for (int r = 0; r < 8; ++r) {
      const int qrow = qw + rbase + r;
      float s0 = sacc[0][r] * scale;
      float s1 = sacc[1][r] * scale;
      if (sb + ml > qrow)      s0 = -1e30f;
      if (sb + 16 + ml > qrow) s1 = -1e30f;
      float rmax = fmaxf(s0, s1);
#pragma unroll
      for (int o = 8; o > 0; o >>= 1) rmax = fmaxf(rmax, __shfl_xor(rmax, o));
      const float mnew = fmaxf(mrow[r], rmax);
      const float fac = __expf(mrow[r] - mnew);
      const float p0 = __expf(s0 - mnew);
      const float p1 = __expf(s1 - mnew);
      float rsum = p0 + p1;
#pragma unroll
      for (int o = 8; o > 0; o >>= 1) rsum += __shfl_xor(rsum, o);
      lrow[r] = lrow[r] * fac + rsum;
      mrow[r] = mnew;
#pragma unroll
      for (int j = 0; j < 4; ++j) oacc[j][r] *= fac;
      const int m = rbase + r;
      P[wave][m][ml]      = (bf16)p0;
      P[wave][m][16 + ml] = (bf16)p1;
    }
    __syncthreads();

    // O += P (16x32) * V (32x64): load P + all 4 V frags, then 4 WMMAs
    v16bf pf = load_frag(&P[wave][ml][lh], &P[wave][ml][lh + 16]);
    v16bf vf[4];
#pragma unroll
    for (int j = 0; j < 4; ++j) {
      const int d = (j << 4) + ml;
      vf[j] = load_frag(&Vt[d][lh], &Vt[d][lh + 16]);
    }
#pragma unroll
    for (int j = 0; j < 4; ++j)
      oacc[j] = WMMA_BF16(pf, vf[j], oacc[j]);
    __syncthreads();
  }

  // write normalized O as bf16 into [B*T, 1024]
#pragma unroll
  for (int r = 0; r < 8; ++r) {
    const float inv = 1.f / lrow[r];
    const int t = qw + rbase + r;
    bf16* op = out + ((size_t)b * T + t) * 1024 + h * 64;
#pragma unroll
    for (int j = 0; j < 4; ++j)
      op[(j << 4) + ml] = (bf16)(oacc[j][r] * inv);
  }
}

// ---------------------------------------------------------------------------
// Orchestration
// ---------------------------------------------------------------------------
extern "C" void kernel_launch(void* const* d_in, const int* in_sizes, int n_in,
                              void* d_out, int out_size, void* d_ws, size_t ws_size,
                              hipStream_t stream) {
  (void)in_sizes; (void)n_in; (void)out_size; (void)ws_size;

  const float* x  = (const float*)d_in[0];
  const float* Wq = (const float*)d_in[1];
  const float* Wk = (const float*)d_in[2];
  const float* Wv = (const float*)d_in[3];
  const float* Wo = (const float*)d_in[4];
  const float* bo = (const float*)d_in[5];
  const float* W1 = (const float*)d_in[6];
  const float* b1 = (const float*)d_in[7];
  const float* W2 = (const float*)d_in[8];
  const float* b2 = (const float*)d_in[9];
  const float* g1  = (const float*)d_in[10];
  const float* be1 = (const float*)d_in[11];
  const float* g2  = (const float*)d_in[12];
  const float* be2 = (const float*)d_in[13];

  const int Bsz = 2, T = 2048, C = 1024, H = 16, DH = 64;
  const int BT = Bsz * T;          // 4096
  const int N3 = 3 * H * DH;       // 3072
  const int F  = 4 * C;            // 4096

  const size_t MB = (size_t)1 << 20;
  char* ws = (char*)d_ws;
  bf16* hbf   = (bf16*)(ws +  0 * MB);  //  8 MB: LN1 out, later LN2 out
  bf16* wqkv  = (bf16*)(ws +  8 * MB);  //  6 MB
  bf16* wo    = (bf16*)(ws + 14 * MB);  //  2 MB
  bf16* w1    = (bf16*)(ws + 16 * MB);  //  8 MB
  bf16* w2    = (bf16*)(ws + 24 * MB);  //  8 MB
  bf16* qkv   = (bf16*)(ws + 32 * MB);  // 24 MB (recycled as ff1 later)
  bf16* ff1   = (bf16*)(ws + 32 * MB);  // 32 MB [32,64)
  bf16* attn  = (bf16*)(ws + 64 * MB);  //  8 MB
  float* x1   = (float*)(ws + 72 * MB); // 16 MB; total 88 MB

  // --- weight prep (bf16) ---
  {
    int n = C * N3;
    pack_qkv_kernel<<<(n + 255) / 256, 256, 0, stream>>>(Wq, Wk, Wv, wqkv, H, C, DH);
  }
  f32_to_bf16_kernel<<<(C * C + 255) / 256, 256, 0, stream>>>(Wo, wo, C * C);
  f32_to_bf16_kernel<<<(C * F + 255) / 256, 256, 0, stream>>>(W1, w1, C * F);
  f32_to_bf16_kernel<<<(F * C + 255) / 256, 256, 0, stream>>>(W2, w2, F * C);

  // --- attention ---
  ln_bf16_kernel<<<BT, 256, 0, stream>>>(x, g1, be1, hbf, C);
  gemm_bf16_kernel<<<dim3(N3 / 128, BT / 128), 256, 0, stream>>>(
      hbf, wqkv, nullptr, nullptr, nullptr, qkv, BT, N3, C, 0);
  attn_kernel<<<dim3(T / 64, Bsz * H), 128, 0, stream>>>(qkv, attn, T);
  gemm_bf16_kernel<<<dim3(C / 128, BT / 128), 256, 0, stream>>>(
      attn, wo, bo, x, x1, nullptr, BT, C, H * DH, 0);

  // --- FFN ---
  ln_bf16_kernel<<<BT, 256, 0, stream>>>(x1, g2, be2, hbf, C);
  gemm_bf16_kernel<<<dim3(F / 128, BT / 128), 256, 0, stream>>>(
      hbf, w1, b1, nullptr, nullptr, ff1, BT, F, C, 1);
  gemm_bf16_kernel<<<dim3(C / 128, BT / 128), 256, 0, stream>>>(
      ff1, w2, b2, x1, (float*)d_out, nullptr, BT, C, F, 0);
}
